// SAGAN_Attention_56547539419342
// MI455X (gfx1250) — matrix operand order repaired and checked by reading, not compiled, not run on hardware
//
#include <hip/hip_runtime.h>

// ---------------- CDNA5 WMMA types ----------------
typedef __attribute__((ext_vector_type(16))) __bf16 bf16x16;
typedef __attribute__((ext_vector_type(8)))  __bf16 bf16x8;
typedef __attribute__((ext_vector_type(8)))  float  f32x8;

static __device__ __forceinline__ f32x8 wmma_bf16(bf16x16 a, bf16x16 b, f32x8 c) {
    // (neg_a, A, neg_b, B, c_mod, C, reuse_a, reuse_b)
    return __builtin_amdgcn_wmma_f32_16x16x32_bf16(false, a, false, b, (short)0, c, false, false);
}

static __device__ __forceinline__ bf16x16 join8(bf16x8 lo, bf16x8 hi) {
    bf16x16 v;
#pragma unroll
    for (int i = 0; i < 8; ++i) { v[i] = lo[i]; v[i + 8] = hi[i]; }
    return v;
}

// Problem constants
#define NB   8
#define NC   256
#define NH   64
#define NW   64
#define NN   4096   // H*W
#define NM   1024   // pooled N/4
#define C8   32
#define C2   128
#define OSUM 192    // 32 + 32 + 128 stacked projection rows

// ---------------- Workspace layout (bytes, 256-aligned) ----------------
// wcat   bf16 [192][256]        :      98304
// woT    bf16 [256][128]        :      65536
// thetaT bf16 [B][4096][32]     :    2097152   ([n][c])
// phiFT  bf16 [B][4096][32]     :    2097152   ([n][c])
// gFT    bf16 [B][4096][128]    :    8388608   ([n][c])
// phiT   bf16 [B][1024][32]     :     524288   ([m][c])
// gP     bf16 [B][128][1024]    :    2097152   ([c][m])
// oT     bf16 [B][4096][128]    :    8388608   ([n][c])
static constexpr size_t OFF_WCAT   = 0;
static constexpr size_t OFF_WOT    = OFF_WCAT   + 98304;
static constexpr size_t OFF_THETAT = OFF_WOT    + 65536;
static constexpr size_t OFF_PHIFT  = OFF_THETAT + 2097152;
static constexpr size_t OFF_GFT    = OFF_PHIFT  + 2097152;
static constexpr size_t OFF_PHIT   = OFF_GFT    + 8388608;
static constexpr size_t OFF_GP     = OFF_PHIT   + 524288;
static constexpr size_t OFF_OT     = OFF_GP     + 2097152;

// ================= Kernel 0: weights -> bf16 =================
__global__ void __launch_bounds__(256)
k_wcvt(const float* __restrict__ wt, const float* __restrict__ wp,
       const float* __restrict__ wg, const float* __restrict__ wo,
       __bf16* __restrict__ wcat, __bf16* __restrict__ woT) {
    int idx = blockIdx.x * blockDim.x + threadIdx.x;
    const int n1 = OSUM * NC;          // 49152
    const int n2 = NC * C2;            // 32768
    if (idx < n1) {
        int row = idx / NC, col = idx % NC;
        float v = (row < 32) ? wt[row * NC + col]
                : (row < 64) ? wp[(row - 32) * NC + col]
                             : wg[(row - 64) * NC + col];
        wcat[idx] = (__bf16)v;
    } else if (idx < n1 + n2) {
        int j = idx - n1;
        woT[j] = (__bf16)wo[j];
    }
}

// ================= Kernel 1: fused projection GEMM =================
// out[o,n] = sum_c wcat[o,c] * x[b,c,n], o in [0,192); block handles 64 n-cols.
__global__ void __launch_bounds__(256)
k_proj(const float* __restrict__ x, const __bf16* __restrict__ wcat,
       __bf16* __restrict__ thetaT, __bf16* __restrict__ phiFT,
       __bf16* __restrict__ gFT) {
    __shared__ __bf16 xT[64 * 256];   // [n_local][c], 32 KB
    const int b  = blockIdx.y;
    const int n0 = blockIdx.x * 64;
    const int tid = threadIdx.x;

    // Stage & transpose the x tile: thread t owns channel c=t (contiguous global reads)
    {
        const int c = tid;
        const float4* p = (const float4*)(x + ((size_t)(b * NC + c)) * NN + n0);
#pragma unroll 4
        for (int k4 = 0; k4 < 16; ++k4) {
            float4 f = p[k4];
            int nl = k4 * 4;
            xT[(nl + 0) * 256 + c] = (__bf16)f.x;
            xT[(nl + 1) * 256 + c] = (__bf16)f.y;
            xT[(nl + 2) * 256 + c] = (__bf16)f.z;
            xT[(nl + 3) * 256 + c] = (__bf16)f.w;
        }
    }
    __syncthreads();

    const int wave = tid >> 5, lane = tid & 31;
    const int r = lane & 15, half = lane >> 4;

    // 12 o-tiles x 4 n-tiles = 48 tiles, 6 per wave (no outer unroll -> no spills)
#pragma unroll 1
    for (int t = 0; t < 6; ++t) {
        const int idx = wave * 6 + t;
        const int ot = idx >> 2, nt = idx & 3;
        const int o0 = ot * 16;
        const __bf16* wrow = wcat + (size_t)(o0 + r) * NC;
        const __bf16* xrow = &xT[(nt * 16 + r) * 256];
        f32x8 acc0 = {}, acc1 = {};
#pragma unroll
        for (int kc = 0; kc < 8; kc += 2) {
            bf16x16 a0 = *(const bf16x16*)(wrow + kc * 32 + 16 * half);
            bf16x16 b0 = *(const bf16x16*)(xrow + kc * 32 + 16 * half);
            acc0 = wmma_bf16(a0, b0, acc0);
            bf16x16 a1 = *(const bf16x16*)(wrow + (kc + 1) * 32 + 16 * half);
            bf16x16 b1 = *(const bf16x16*)(xrow + (kc + 1) * 32 + 16 * half);
            acc1 = wmma_bf16(a1, b1, acc1);
        }
        f32x8 acc = acc0 + acc1;

        // Pack 8 consecutive output channels -> one 16B store (uniform per-tile branch)
        bf16x8 pk;
#pragma unroll
        for (int i = 0; i < 8; ++i) pk[i] = (__bf16)acc[i];
        const int gn = n0 + nt * 16 + r;
        if (ot < 2) {
            *(bf16x8*)(thetaT + ((size_t)b * NN + gn) * C8 + o0 + 8 * half) = pk;
        } else if (ot < 4) {
            *(bf16x8*)(phiFT + ((size_t)b * NN + gn) * C8 + (o0 - 32) + 8 * half) = pk;
        } else {
            *(bf16x8*)(gFT + ((size_t)b * NN + gn) * C2 + (o0 - 64) + 8 * half) = pk;
        }
    }
}

// ================= Kernel 2: 2x2 maxpool into WMMA-friendly layouts =================
__global__ void __launch_bounds__(256)
k_pool(const __bf16* __restrict__ phiFT, const __bf16* __restrict__ gFT,
       __bf16* __restrict__ phiT, __bf16* __restrict__ gP) {
    int gid = blockIdx.x * blockDim.x + threadIdx.x;
    const int nPhi = NB * NM * C8;   // 262144
    const int nG   = NB * NM * C2;   // 1048576
    if (gid < nPhi) {
        const int b = gid / (NM * C8);
        const int rem = gid % (NM * C8);
        const int m = rem / C8, c = rem % C8;
        const int ph = m >> 5, pw = m & 31;
        const int n00 = ph * 2 * NW + pw * 2;
        const __bf16* s = phiFT + (size_t)b * NN * C8;
        float v = fmaxf(fmaxf((float)s[(n00) * C8 + c], (float)s[(n00 + 1) * C8 + c]),
                        fmaxf((float)s[(n00 + NW) * C8 + c], (float)s[(n00 + NW + 1) * C8 + c]));
        phiT[((size_t)b * NM + m) * C8 + c] = (__bf16)v;
    } else if (gid < nPhi + nG) {
        const int g2 = gid - nPhi;
        const int b = g2 / (NM * C2);
        const int rem = g2 % (NM * C2);
        const int c = rem / NM, m = rem % NM;
        const int ph = m >> 5, pw = m & 31;
        const int n00 = ph * 2 * NW + pw * 2;
        const __bf16* s = gFT + (size_t)b * NN * C2;
        float v = fmaxf(fmaxf((float)s[(n00) * C2 + c], (float)s[(n00 + 1) * C2 + c]),
                        fmaxf((float)s[(n00 + NW) * C2 + c], (float)s[(n00 + NW + 1) * C2 + c]));
        gP[((size_t)b * C2 + c) * NM + m] = (__bf16)v;
    }
}

// ================= Kernel 3: fused attention (logits + softmax + o) =================
// One block (8 waves) per 16-row n-tile. S strip kept in LDS; beta never hits HBM.
__global__ void __launch_bounds__(256)
k_attn(const __bf16* __restrict__ thetaT, const __bf16* __restrict__ phiT,
       const __bf16* __restrict__ gP, __bf16* __restrict__ oT) {
    extern __shared__ char smem[];
    float*  sS   = (float*)smem;                                  // [16][1024] f32
    __bf16* sP   = (__bf16*)(smem + 16 * 1024 * 4);               // [16][1024] bf16
    float*  sMax = (float*)(smem + 16 * 1024 * 4 + 16 * 1024 * 2);// [16][16]
    float*  sSum = sMax + 256;                                    // [16][16]

    const int b  = blockIdx.y;
    const int n0 = blockIdx.x * 16;
    const int tid = threadIdx.x;
    const int wave = tid >> 5, lane = tid & 31;
    const int r = lane & 15, half = lane >> 4;

    // A fragment: theta^T rows (shared by all 8 m-tiles of this wave)
    const __bf16* trow = thetaT + ((size_t)b * NN + n0 + r) * C8;
    bf16x16 aT = *(const bf16x16*)(trow + 16 * half);

    // ---- Phase 1: S[n_tile, 0..1023]; wave owns 128 keys ----
#pragma unroll 2
    for (int t = 0; t < 8; ++t) {
        const int m0 = wave * 128 + t * 16;
        bf16x16 bb = *(const bf16x16*)(phiT + ((size_t)b * NM + m0 + r) * C8 + 16 * half);
        f32x8 acc = {};
        acc = wmma_bf16(aT, bb, acc);     // single K=32 covers C/8
#pragma unroll
        for (int i = 0; i < 8; ++i)
            sS[(i + 8 * half) * NM + m0 + r] = acc[i];
    }
    __syncthreads();

    // ---- Softmax over 1024 keys: 16 threads per row ----
    const int tr = tid >> 4, tc = tid & 15;
    float* row = sS + tr * NM;
    float mx = -3.4e38f;
#pragma unroll 8
    for (int k = 0; k < 64; ++k) mx = fmaxf(mx, row[tc + 16 * k]);
    sMax[tr * 16 + tc] = mx;
    __syncthreads();
    mx = sMax[tr * 16];
#pragma unroll
    for (int k = 1; k < 16; ++k) mx = fmaxf(mx, sMax[tr * 16 + k]);
    float sum = 0.f;
#pragma unroll 8
    for (int k = 0; k < 64; ++k) {
        float e = __expf(row[tc + 16 * k] - mx);
        row[tc + 16 * k] = e;
        sum += e;
    }
    sSum[tr * 16 + tc] = sum;
    __syncthreads();
    sum = 0.f;
#pragma unroll
    for (int k = 0; k < 16; ++k) sum += sSum[tr * 16 + k];
    const float inv = 1.0f / sum;
#pragma unroll 8
    for (int k = 0; k < 64; ++k)
        sP[tr * NM + tc + 16 * k] = (__bf16)(row[tc + 16 * k] * inv);
    __syncthreads();

    // ---- Phase 2: oT[n_tile, c_tile] = p (16x1024) x g^T; wave owns 16 channels ----
    const int c0 = wave * 16;
    const __bf16* gRow = gP + ((size_t)b * C2 + c0 + r) * NM;
    const __bf16* pRow = sP + r * NM;
    f32x8 acc0 = {}, acc1 = {};
#pragma unroll 2
    for (int kt = 0; kt < 32; kt += 2) {
        bf16x16 a0 = *(const bf16x16*)(pRow + kt * 32 + 16 * half);
        bf16x16 b0 = *(const bf16x16*)(gRow + kt * 32 + 16 * half);
        acc0 = wmma_bf16(a0, b0, acc0);
        bf16x16 a1 = *(const bf16x16*)(pRow + (kt + 1) * 32 + 16 * half);
        bf16x16 b1 = *(const bf16x16*)(gRow + (kt + 1) * 32 + 16 * half);
        acc1 = wmma_bf16(a1, b1, acc1);
    }
    f32x8 acc = acc0 + acc1;

    // Stage o tile through (now dead) sS region, then packed 16B stores
    float* sOut = sS;                 // [16][128] f32, reuse
#pragma unroll
    for (int i = 0; i < 8; ++i)
        sOut[(i + 8 * half) * C2 + c0 + r] = acc[i];
    __syncthreads();
    {
        const int n = tid >> 4;
        const int cc = (tid & 15) * 8;
        bf16x8 pk;
#pragma unroll
        for (int j = 0; j < 8; ++j) pk[j] = (__bf16)sOut[n * C2 + cc + j];
        *(bf16x8*)(oT + ((size_t)b * NN + n0 + n) * C2 + cc) = pk;
    }
}

// ================= Kernel 4: output projection + residual =================
__global__ void __launch_bounds__(256)
k_out(const float* __restrict__ x, const __bf16* __restrict__ woT,
      const __bf16* __restrict__ oTbuf, const float* __restrict__ gamma,
      float* __restrict__ out) {
    __shared__ float sTile[8][16 * 16];   // per-wave transpose buffer, 8 KB
    const int b  = blockIdx.y;
    const int n0 = blockIdx.x * 64;
    const int tid = threadIdx.x;
    const int wave = tid >> 5, lane = tid & 31;
    const int r = lane & 15, half = lane >> 4;
    const float gm = gamma[0];
    float* st = sTile[wave];

    // 16 oc-tiles x 4 n-tiles = 64 tiles, 8 per wave
#pragma unroll 1
    for (int t = 0; t < 8; ++t) {
        const int idx = wave * 8 + t;
        const int ot = idx >> 2, nt = idx & 3;
        const int oc0 = ot * 16;
        const int nbase = n0 + nt * 16;
        const __bf16* wrow = woT + (size_t)(oc0 + r) * C2;
        const __bf16* orow = oTbuf + ((size_t)b * NN + nbase + r) * C2;
        f32x8 acc0 = {}, acc1 = {};
#pragma unroll
        for (int kc = 0; kc < 4; kc += 2) {
            bf16x16 a0 = *(const bf16x16*)(wrow + kc * 32 + 16 * half);
            bf16x16 b0 = *(const bf16x16*)(orow + kc * 32 + 16 * half);
            acc0 = wmma_bf16(a0, b0, acc0);
            bf16x16 a1 = *(const bf16x16*)(wrow + (kc + 1) * 32 + 16 * half);
            bf16x16 b1 = *(const bf16x16*)(orow + (kc + 1) * 32 + 16 * half);
            acc1 = wmma_bf16(a1, b1, acc1);
        }
        f32x8 acc = acc0 + acc1;

        // Transpose tile through per-wave LDS (same-wave LDS ops are in order)
#pragma unroll
        for (int i = 0; i < 8; ++i)
            st[(i + 8 * half) * 16 + r] = acc[i];
#pragma unroll
        for (int q = 0; q < 2; ++q) {
            const int fid = lane + 32 * q;     // 0..63 float4s
            const int row = fid >> 2, col4 = fid & 3;
            float4 v = *(float4*)&st[row * 16 + col4 * 4];
            const size_t oi = ((size_t)(b * NC + oc0 + row)) * NN + (nbase + col4 * 4);
            float4 xr = *(const float4*)(x + oi);
            float4 o;
            o.x = gm * v.x + xr.x; o.y = gm * v.y + xr.y;
            o.z = gm * v.z + xr.z; o.w = gm * v.w + xr.w;
            *(float4*)(out + oi) = o;
        }
    }
}

// ================= Host launcher =================
extern "C" void kernel_launch(void* const* d_in, const int* in_sizes, int n_in,
                              void* d_out, int out_size, void* d_ws, size_t ws_size,
                              hipStream_t stream) {
    (void)in_sizes; (void)n_in; (void)out_size; (void)ws_size;
    const float* x       = (const float*)d_in[0];
    const float* w_theta = (const float*)d_in[1];
    const float* w_phi   = (const float*)d_in[2];
    const float* w_g     = (const float*)d_in[3];
    const float* w_o     = (const float*)d_in[4];
    const float* gamma   = (const float*)d_in[5];
    float* out = (float*)d_out;

    char* ws = (char*)d_ws;
    __bf16* wcat   = (__bf16*)(ws + OFF_WCAT);
    __bf16* woT    = (__bf16*)(ws + OFF_WOT);
    __bf16* thetaT = (__bf16*)(ws + OFF_THETAT);
    __bf16* phiFT  = (__bf16*)(ws + OFF_PHIFT);
    __bf16* gFT    = (__bf16*)(ws + OFF_GFT);
    __bf16* phiT   = (__bf16*)(ws + OFF_PHIT);
    __bf16* gP     = (__bf16*)(ws + OFF_GP);
    __bf16* oT     = (__bf16*)(ws + OFF_OT);

    // 0) weight conversion
    {
        const int tot = OSUM * NC + NC * C2;
        k_wcvt<<<(tot + 255) / 256, 256, 0, stream>>>(w_theta, w_phi, w_g, w_o, wcat, woT);
    }
    // 1) fused projection GEMM
    k_proj<<<dim3(NN / 64, NB), 256, 0, stream>>>(x, wcat, thetaT, phiFT, gFT);
    // 2) maxpool + layout
    {
        const int tot = NB * NM * (C8 + C2);
        k_pool<<<(tot + 255) / 256, 256, 0, stream>>>(phiFT, gFT, phiT, gP);
    }
    // 3) fused attention (100 KB dynamic LDS; 320 KB/WGP on CDNA5)
    {
        const int smemBytes = 16 * 1024 * 4 + 16 * 1024 * 2 + 2 * 256 * 4; // 100352
        hipFuncSetAttribute((const void*)k_attn,
                            hipFuncAttributeMaxDynamicSharedMemorySize, smemBytes);
        k_attn<<<dim3(NN / 16, NB), 256, smemBytes, stream>>>(thetaT, phiT, gP, oT);
    }
    // 4) output projection + residual
    k_out<<<dim3(NN / 64, NB), 256, 0, stream>>>(x, woT, oT, gamma, out);
}
